// RelMarkerv0_5411658793028
// MI455X (gfx1250) — matrix operand order repaired and checked by reading, not compiled, not run on hardware
//
#include <hip/hip_runtime.h>
#include <hip/hip_bf16.h>
#include <stdint.h>

// Problem dims (fixed by the reference)
#define BATCH 4
#define SEQ   512
#define DIM   768
#define KSP   96
#define HID   3072
#define NPAIR (KSP * (KSP - 1))   // 9120 pairs per batch
#define ROWS  (BATCH * KSP)       // 384 gathered token rows
#define PROWS (BATCH * NPAIR)     // 36480 total pair rows

// Pair-kernel tiling: 64 rows x 256 cols per block, 8 waves as 4(M) x 2(N),
// each wave: 16 rows x 8 strips of 16 cols.
#define MTILE 64
#define NBLK  256
#define WSTRIP 8

typedef __attribute__((ext_vector_type(16))) __bf16 v16bf;
typedef __attribute__((ext_vector_type(8)))  float  v8f;
typedef __attribute__((ext_vector_type(4)))  unsigned int v4u;

union FragU { v16bf v; v4u q[2]; };

// round-to-nearest-even f32 -> bf16 bits (one-time repack paths)
__device__ __forceinline__ unsigned short f2bf(float f) {
    unsigned u = __float_as_uint(f);
    unsigned r = u + 0x7FFFu + ((u >> 16) & 1u);
    return (unsigned short)(r >> 16);
}

// cheap packed f32x2 -> bf16x2: add-round then one v_perm_b32 picking both high halves
__device__ __forceinline__ unsigned pack_bf16x2(float lo, float hi) {
    unsigned tl = __float_as_uint(lo) + 0x8000u;
    unsigned th = __float_as_uint(hi) + 0x8000u;
    return __builtin_amdgcn_perm(th, tl, 0x07060302u);
}

// A-matrix 16x32 bf16 fragment: lanes 0-15 -> row M=lane&15, K=kb..kb+15
// with kb = (lane>>4)*16 (matches load_frag_b).  Two contiguous 16B loads per lane.
__device__ __forceinline__ v16bf load_frag_a(const unsigned short* base, int ld, int lane) {
    const unsigned short* p = base + (long long)(lane & 15) * ld + ((lane >> 4) << 4);
    FragU f;
    f.q[0] = *(const v4u*)(p);
    f.q[1] = *(const v4u*)(p + 8);
    return f.v;
}

// B-matrix 32x16 bf16 fragment: lane holds column N=lane&15; lanes 0-15 K=0..15,
// lanes 16-31 K=16..31.  Storage is B-transposed [N][K] -> one 32B contiguous read.
__device__ __forceinline__ v16bf load_frag_b(const unsigned short* base, int ld, int lane) {
    const unsigned short* p = base + (long long)(lane & 15) * ld + ((lane >> 4) << 4);
    FragU f;
    f.q[0] = *(const v4u*)(p);
    f.q[1] = *(const v4u*)(p + 8);
    return f.v;
}

// ---------------------------------------------------------------------------
// Weight repack: f32 [K][N] row-major  ->  bf16 [N][K]  (coalesced writes)
// ---------------------------------------------------------------------------
__global__ __launch_bounds__(256) void transpose_convert_kernel(
    const float* __restrict__ W, unsigned short* __restrict__ Wt, int Kd, int Nd)
{
    int idx = blockIdx.x * 256 + threadIdx.x;
    int total = Kd * Nd;
    if (idx >= total) return;
    int n = idx / Kd;
    int k = idx - n * Kd;
    Wt[idx] = f2bf(W[(long long)k * Nd + n]);
}

// ---------------------------------------------------------------------------
// Gather the K start / end token rows of h and convert to bf16
// ---------------------------------------------------------------------------
__global__ __launch_bounds__(256) void gather_convert_kernel(
    const float* __restrict__ h, const int* __restrict__ span,
    unsigned short* __restrict__ hs, unsigned short* __restrict__ he)
{
    int r = blockIdx.x;               // 0..383
    int b = r / KSP;
    int si = span[r * 2 + 0];
    int ei = span[r * 2 + 1];
    const float* ps = h + (long long)(b * SEQ + si) * DIM;
    const float* pe = h + (long long)(b * SEQ + ei) * DIM;
    for (int d = threadIdx.x; d < DIM; d += blockDim.x) {
        hs[(long long)r * DIM + d] = f2bf(ps[d]);
        he[(long long)r * DIM + d] = f2bf(pe[d]);
    }
}

// ---------------------------------------------------------------------------
// Generic bf16 WMMA GEMM: C[M,N] = op(A[M,K] @ B[K,N] + bias)
// A bf16 row-major (lda), B stored transposed bf16 [N][K] (ldbt).
// Block = 256 threads = 8 waves; tile = 16 rows x 128 cols (16x16 per wave).
// ---------------------------------------------------------------------------
template<int RELU, int OUTBF>
__global__ __launch_bounds__(256) void gemm_bf16_kernel(
    const unsigned short* __restrict__ A, int lda,
    const unsigned short* __restrict__ Bt, int ldbt,
    const float* __restrict__ bias,
    void* __restrict__ Cout, int ldc, int Kdim)
{
    const int lane = threadIdx.x & 31;
    const int wave = threadIdx.x >> 5;
    const int n0 = blockIdx.x * 128 + wave * 16;
    const int m0 = blockIdx.y * 16;
    const unsigned short* Ab = A + (long long)m0 * lda;
    const unsigned short* Bb = Bt + (long long)n0 * ldbt;

    v8f acc = {0.f, 0.f, 0.f, 0.f, 0.f, 0.f, 0.f, 0.f};
    for (int k = 0; k < Kdim; k += 32) {
        v16bf a = load_frag_a(Ab + k, lda, lane);
        v16bf b = load_frag_b(Bb + k, ldbt, lane);
        acc = __builtin_amdgcn_wmma_f32_16x16x32_bf16(false, a, false, b,
                                                      (short)0, acc, false, false);
    }

    const int n  = n0 + (lane & 15);
    const int mr = m0 + ((lane >> 4) << 3);
    const float bv = bias ? bias[n] : 0.0f;
#pragma unroll
    for (int v = 0; v < 8; ++v) {
        float x = acc[v] + bv;
        if (RELU) x = fmaxf(x, 0.0f);
        if (OUTBF) ((unsigned short*)Cout)[(long long)(mr + v) * ldc + n] = f2bf(x);
        else       ((float*)Cout)[(long long)(mr + v) * ldc + n] = x;
    }
}

// ---------------------------------------------------------------------------
// Pair kernel: pair-row g -> (b,i,j); hidden = relu(U[b,i,:] + V[b,j,:])
// (ro_b1 pre-folded into U); out = hidden @ ro_w2 + ro_b2.
//
// Block = 256 thr / 8 waves computes a 64 x 256 output tile.  Waves are
// arranged 4(M) x 2(N): wave wm owns 16 rows, wave wn owns 128 cols as 8
// strips of 16 -> 64 acc VGPRs, 16 WMMAs per 64-K chunk per wave.
// 64-row A chunks are built in double-buffered LDS (relu(U_i+V_j) -> bf16),
// one barrier per chunk; B fragments stream from the L2-resident ro_w2^T,
// read once per 64-row group (4x less L2 traffic than a 16-row tile).
// ---------------------------------------------------------------------------
__global__ __launch_bounds__(256) void pair_gemm_kernel(
    const float* __restrict__ U, const float* __restrict__ V,
    const unsigned short* __restrict__ W2t, const float* __restrict__ b2,
    float* __restrict__ out)
{
    __shared__ unsigned short As[2][MTILE * 72];   // 2 x (64 rows x 64 k), stride 72 (144B)

    const int lane = threadIdx.x & 31;
    const int wave = threadIdx.x >> 5;
    const int wm = wave & 3;                 // M sub-tile (16 rows each)
    const int wn = wave >> 2;                // N half (128 cols each)
    const int m0 = blockIdx.x * MTILE;       // pair-row base
    const int n0base = blockIdx.y * NBLK + wn * 128;

    // staging: thread t fills row t>>2 (0..63), 16 consecutive k at (t&3)*16
    const int srow = threadIdx.x >> 2;
    const int scol = (threadIdx.x & 3) << 4;
    int g  = m0 + srow;
    int b  = g / NPAIR;
    int p  = g - b * NPAIR;
    int i  = p / (KSP - 1);
    int jr = p - i * (KSP - 1);
    int j  = jr + (jr >= i ? 1 : 0);
    const float* Urow = U + (long long)(b * KSP + i) * HID + scol;
    const float* Vrow = V + (long long)(b * KSP + j) * HID + scol;

    v8f acc[WSTRIP];
#pragma unroll
    for (int s = 0; s < WSTRIP; ++s)
        acc[s] = (v8f){0.f, 0.f, 0.f, 0.f, 0.f, 0.f, 0.f, 0.f};

    auto stage = [&](int buf, int kc) {
        unsigned* dst = (unsigned*)&As[buf][srow * 72 + scol];
#pragma unroll
        for (int q = 0; q < 4; ++q) {
            const float4 u4 = *(const float4*)(Urow + kc + 4 * q);
            const float4 v4 = *(const float4*)(Vrow + kc + 4 * q);
            dst[2 * q + 0] = pack_bf16x2(fmaxf(u4.x + v4.x, 0.f), fmaxf(u4.y + v4.y, 0.f));
            dst[2 * q + 1] = pack_bf16x2(fmaxf(u4.z + v4.z, 0.f), fmaxf(u4.w + v4.w, 0.f));
        }
    };

    stage(0, 0);
    int cur = 0;
    for (int kc = 0; kc < HID; kc += 64) {
        __syncthreads();                      // staged chunk `cur` visible; prior reads done
        const unsigned short* Arows = &As[cur][(wm * 16) * 72];
        v16bf a0 = load_frag_a(Arows,      72, lane);
        v16bf a1 = load_frag_a(Arows + 32, 72, lane);
        if (kc + 64 < HID) stage(cur ^ 1, kc + 64);
#pragma unroll
        for (int s = 0; s < WSTRIP; ++s) {
            const unsigned short* Bs = W2t + (long long)(n0base + s * 16) * HID + kc;
            v16bf b0 = load_frag_b(Bs,      HID, lane);
            v16bf b1 = load_frag_b(Bs + 32, HID, lane);
            acc[s] = __builtin_amdgcn_wmma_f32_16x16x32_bf16(false, a0, false, b0,
                                                             (short)0, acc[s], false, false);
            acc[s] = __builtin_amdgcn_wmma_f32_16x16x32_bf16(false, a1, false, b1,
                                                             (short)0, acc[s], false, false);
        }
        cur ^= 1;
    }

    const int mr = m0 + wm * 16 + ((lane >> 4) << 3);
#pragma unroll
    for (int s = 0; s < WSTRIP; ++s) {
        const int n = n0base + s * 16 + (lane & 15);
        const float bv = b2[n];
#pragma unroll
        for (int v = 0; v < 8; ++v) {
            out[(long long)(mr + v) * DIM + n] = acc[s][v] + bv;
        }
    }
}

// ---------------------------------------------------------------------------
extern "C" void kernel_launch(void* const* d_in, const int* in_sizes, int n_in,
                              void* d_out, int out_size, void* d_ws, size_t ws_size,
                              hipStream_t stream)
{
    (void)in_sizes; (void)n_in; (void)out_size; (void)ws_size;
    const float* h     = (const float*)d_in[0];
    const int*   span  = (const int*)d_in[1];
    const float* ps_w1 = (const float*)d_in[2];
    const float* ps_b1 = (const float*)d_in[3];
    const float* ps_w2 = (const float*)d_in[4];
    const float* ps_b2 = (const float*)d_in[5];
    const float* pe_w1 = (const float*)d_in[6];
    const float* pe_b1 = (const float*)d_in[7];
    const float* pe_w2 = (const float*)d_in[8];
    const float* pe_b2 = (const float*)d_in[9];
    const float* so_w1 = (const float*)d_in[10];
    const float* so_b1 = (const float*)d_in[11];
    const float* so_w2 = (const float*)d_in[12];
    const float* so_b2 = (const float*)d_in[13];
    const float* ro_w1 = (const float*)d_in[14];
    const float* ro_b1 = (const float*)d_in[15];
    const float* ro_w2 = (const float*)d_in[16];
    const float* ro_b2 = (const float*)d_in[17];

    // workspace carve-out (256B aligned)
    char* wsp = (char*)d_ws;
    auto alloc = [&](size_t bytes) -> char* {
        char* q = wsp; wsp += (bytes + 255) & ~(size_t)255; return q;
    };
    unsigned short* ps_w1t = (unsigned short*)alloc((size_t)DIM * HID * 2);
    unsigned short* ps_w2t = (unsigned short*)alloc((size_t)HID * DIM * 2);
    unsigned short* pe_w1t = (unsigned short*)alloc((size_t)DIM * HID * 2);
    unsigned short* pe_w2t = (unsigned short*)alloc((size_t)HID * DIM * 2);
    unsigned short* so_w1t = (unsigned short*)alloc((size_t)2 * DIM * HID * 2);
    unsigned short* so_w2t = (unsigned short*)alloc((size_t)HID * DIM * 2);
    unsigned short* ro_w1t = (unsigned short*)alloc((size_t)2 * DIM * HID * 2);
    unsigned short* ro_w2t = (unsigned short*)alloc((size_t)HID * DIM * 2);
    unsigned short* hs     = (unsigned short*)alloc((size_t)ROWS * DIM * 2);
    unsigned short* he     = (unsigned short*)alloc((size_t)ROWS * DIM * 2);
    unsigned short* hidden = (unsigned short*)alloc((size_t)ROWS * HID * 2);
    unsigned short* cat    = (unsigned short*)alloc((size_t)ROWS * 2 * DIM * 2);
    unsigned short* entity = (unsigned short*)alloc((size_t)ROWS * DIM * 2);
    float*          Ubuf   = (float*)alloc((size_t)ROWS * HID * 4);
    float*          Vbuf   = (float*)alloc((size_t)ROWS * HID * 4);

    auto tlaunch = [&](const float* W, unsigned short* Wt, int Kd, int Nd) {
        int total = Kd * Nd;
        transpose_convert_kernel<<<(total + 255) / 256, 256, 0, stream>>>(W, Wt, Kd, Nd);
    };
    tlaunch(ps_w1, ps_w1t, DIM, HID);
    tlaunch(ps_w2, ps_w2t, HID, DIM);
    tlaunch(pe_w1, pe_w1t, DIM, HID);
    tlaunch(pe_w2, pe_w2t, HID, DIM);
    tlaunch(so_w1, so_w1t, 2 * DIM, HID);
    tlaunch(so_w2, so_w2t, HID, DIM);
    tlaunch(ro_w1, ro_w1t, 2 * DIM, HID);
    tlaunch(ro_w2, ro_w2t, HID, DIM);

    gather_convert_kernel<<<ROWS, 256, 0, stream>>>(h, span, hs, he);

    const dim3 blk(256);
    // start proj: hidden = relu(hs @ ps_w1 + b1); cat[:, :768] = relu(hidden @ ps_w2 + b2)
    gemm_bf16_kernel<1, 1><<<dim3(HID / 128, ROWS / 16), blk, 0, stream>>>(
        hs, DIM, ps_w1t, DIM, ps_b1, hidden, HID, DIM);
    gemm_bf16_kernel<1, 1><<<dim3(DIM / 128, ROWS / 16), blk, 0, stream>>>(
        hidden, HID, ps_w2t, HID, ps_b2, cat, 2 * DIM, HID);
    // end proj: hidden = relu(he @ pe_w1 + b1); cat[:, 768:] = relu(hidden @ pe_w2 + b2)
    gemm_bf16_kernel<1, 1><<<dim3(HID / 128, ROWS / 16), blk, 0, stream>>>(
        he, DIM, pe_w1t, DIM, pe_b1, hidden, HID, DIM);
    gemm_bf16_kernel<1, 1><<<dim3(DIM / 128, ROWS / 16), blk, 0, stream>>>(
        hidden, HID, pe_w2t, HID, pe_b2, cat + DIM, 2 * DIM, HID);
    // entity proj: hidden = relu(cat @ so_w1 + b1); entity = hidden @ so_w2 + b2
    gemm_bf16_kernel<1, 1><<<dim3(HID / 128, ROWS / 16), blk, 0, stream>>>(
        cat, 2 * DIM, so_w1t, 2 * DIM, so_b1, hidden, HID, 2 * DIM);
    gemm_bf16_kernel<0, 1><<<dim3(DIM / 128, ROWS / 16), blk, 0, stream>>>(
        hidden, HID, so_w2t, HID, so_b2, entity, DIM, HID);
    // factored pair layer-1: U = E @ ro_w1[:768] + ro_b1 ; V = E @ ro_w1[768:]
    gemm_bf16_kernel<0, 0><<<dim3(HID / 128, ROWS / 16), blk, 0, stream>>>(
        entity, DIM, ro_w1t, 2 * DIM, ro_b1, Ubuf, HID, DIM);
    gemm_bf16_kernel<0, 0><<<dim3(HID / 128, ROWS / 16), blk, 0, stream>>>(
        entity, DIM, ro_w1t + DIM, 2 * DIM, nullptr, Vbuf, HID, DIM);

    // dominant GEMM: relu(U_i + V_j) @ ro_w2 + b2  -> [4, 9120, 768] f32
    pair_gemm_kernel<<<dim3(PROWS / MTILE, DIM / NBLK), blk, 0, stream>>>(
        Ubuf, Vbuf, ro_w2t, ro_b2, (float*)d_out);
}
